// TemporalTransformer_50173807952525
// MI455X (gfx1250) — matrix operand order repaired and checked by reading, not compiled, not run on hardware
//
#include <hip/hip_runtime.h>
#include <hip/hip_bf16.h>

// ---------------------------------------------------------------------------
// Types for CDNA5 WMMA (wave32): v_wmma_f32_16x16x32_bf16
// ---------------------------------------------------------------------------
typedef __attribute__((ext_vector_type(16))) __bf16 v16bf;
typedef __attribute__((ext_vector_type(8)))  float  v8f;
typedef __attribute__((ext_vector_type(4)))  int    v4i;
typedef unsigned short u16;

union Frag16 { uint4 u[2]; v16bf v; };   // 16 bf16 = 32 bytes = 8 VGPRs
union AccF   { v8f v; float f[8]; };     // 16x16 f32 C/D tile slice per lane

__device__ __forceinline__ u16 f32_to_bf16(float f) {
  unsigned int u = __float_as_uint(f);
  u += 0x7FFFu + ((u >> 16) & 1u);       // round to nearest even
  return (u16)(u >> 16);
}

// ---------------------------------------------------------------------------
// Async global->LDS copy (CDNA5 GLOBAL_LOAD_ASYNC_TO_LDS_B128, ASYNCcnt).
// Signature (from hipcc diagnostic): (int4 as1*, int4 as3*, imm offset, imm cpol)
// Guarded: falls back to a VGPR bounce if this toolchain lacks the builtins.
// ---------------------------------------------------------------------------
#if defined(__has_builtin)
#if __has_builtin(__builtin_amdgcn_global_load_async_to_lds_b128) && \
    __has_builtin(__builtin_amdgcn_s_wait_asynccnt)
#define ATHENA_ASYNC_LDS 1
#endif
#endif
#ifndef ATHENA_ASYNC_LDS
#define ATHENA_ASYNC_LDS 0
#endif

typedef __attribute__((address_space(1))) v4i gas_v4i;  // global
typedef __attribute__((address_space(3))) v4i las_v4i;  // LDS

__device__ __forceinline__ void copy16_g2l(const u16* g, u16* l) {
#if ATHENA_ASYNC_LDS
  __builtin_amdgcn_global_load_async_to_lds_b128(
      (gas_v4i*)g, (las_v4i*)l, /*offset=*/0, /*cpol=*/0);
#else
  *reinterpret_cast<uint4*>(l) = *reinterpret_cast<const uint4*>(g);
#endif
}

__device__ __forceinline__ void wait_async_copies() {
#if ATHENA_ASYNC_LDS
  __builtin_amdgcn_s_wait_asynccnt(0);
#endif
}

// ---------------------------------------------------------------------------
// Conversions / transposes
// ---------------------------------------------------------------------------
__global__ __launch_bounds__(256) void cvt_bf16_kernel(
    const float* __restrict__ in, u16* __restrict__ out, int n) {
  int i = blockIdx.x * 256 + threadIdx.x;
  if (i < n) out[i] = f32_to_bf16(in[i]);
}

// W[N,K] fp32 (row-major) -> Wt[K,N] bf16
__global__ __launch_bounds__(256) void tpose_cvt_kernel(
    const float* __restrict__ W, u16* __restrict__ Wt, int N, int K) {
  int i = blockIdx.x * 256 + threadIdx.x;
  if (i >= N * K) return;
  int n = i / K, k = i - n * K;
  Wt[(size_t)k * N + n] = f32_to_bf16(W[i]);
}

// Kb [B*L, H*64] bf16 -> KbT [(b*H+h)*64+d, L] bf16 (so QK^T B-frags are contiguous)
__global__ __launch_bounds__(256) void kt_tpose_kernel(
    const u16* __restrict__ Kb, u16* __restrict__ KbT) {
  int i = blockIdx.x * 256 + threadIdx.x;     // over 8192*512
  if (i >= 8192 * 512) return;
  int row = i >> 9, col = i & 511;
  int b = row >> 10, l = row & 1023;
  int h = col >> 6,  d = col & 63;
  KbT[(size_t)(((b * 8 + h) * 64) + d) * 1024 + l] = Kb[i];
}

// ---------------------------------------------------------------------------
// Tiled bf16 GEMM: C[M,N] = A[M,K] @ B[K,N] + bias, optional ReLU.
// A,B bf16 row-major; C fp32 (C32) or bf16 (C16). Block 256 thr = 8 waves,
// tile 128x128, K-step 32, each wave: 2x4 WMMA 16x16 tiles.
// Double-buffered LDS with async global->LDS copies: tile i+1 streams into
// LDS (ASYNCcnt) while WMMAs consume tile i.
// ---------------------------------------------------------------------------
__global__ __launch_bounds__(256) void gemm_bf16_kernel(
    const u16* __restrict__ A, const u16* __restrict__ B,
    const float* __restrict__ bias, float* __restrict__ C32,
    u16* __restrict__ C16, int M, int N, int K, int relu) {
  __shared__ u16 As[2][128 * 32];
  __shared__ u16 Bs[2][32 * 128];

  const int tid  = threadIdx.x;
  const int lane = tid & 31;
  const int wave = tid >> 5;
  const int wm   = wave >> 1;          // 0..3  -> 32-row group
  const int wn   = wave & 1;           // 0..1  -> 64-col group
  const int m0   = blockIdx.y * 128;
  const int n0   = blockIdx.x * 128;
  const int mfr  = lane & 15;
  const int kb   = (lane >= 16) ? 8 : 0;

  // This thread's two 16B staging chunks (fixed across K-steps)
  const int c0 = tid,        c1 = tid + 256;
  const int ar0 = c0 >> 2,   ac0 = (c0 & 3) * 8;
  const int ar1 = c1 >> 2,   ac1 = (c1 & 3) * 8;
  const int br0 = c0 >> 4,   bc0 = (c0 & 15) * 8;
  const int br1 = c1 >> 4,   bc1 = (c1 & 15) * 8;

  AccF acc[2][4];
#pragma unroll
  for (int rt = 0; rt < 2; ++rt)
#pragma unroll
    for (int ct = 0; ct < 4; ++ct)
#pragma unroll
      for (int r = 0; r < 8; ++r) acc[rt][ct].f[r] = 0.f;

  const int nstep = K >> 5;

  // Prologue: stage tile 0 into buffer 0
  copy16_g2l(&A[(size_t)(m0 + ar0) * K + ac0], &As[0][ar0 * 32 + ac0]);
  copy16_g2l(&A[(size_t)(m0 + ar1) * K + ac1], &As[0][ar1 * 32 + ac1]);
  copy16_g2l(&B[(size_t)br0 * N + n0 + bc0],   &Bs[0][br0 * 128 + bc0]);
  copy16_g2l(&B[(size_t)br1 * N + n0 + bc1],   &Bs[0][br1 * 128 + bc1]);
  wait_async_copies();
  __syncthreads();

  for (int step = 0; step < nstep; ++step) {
    const int cur = step & 1;
    const int k0  = step << 5;

    if (step + 1 < nstep) {  // stream next tile into the other buffer (async)
      const int nxt = cur ^ 1;
      const int kn  = k0 + 32;
      copy16_g2l(&A[(size_t)(m0 + ar0) * K + kn + ac0], &As[nxt][ar0 * 32 + ac0]);
      copy16_g2l(&A[(size_t)(m0 + ar1) * K + kn + ac1], &As[nxt][ar1 * 32 + ac1]);
      copy16_g2l(&B[(size_t)(kn + br0) * N + n0 + bc0], &Bs[nxt][br0 * 128 + bc0]);
      copy16_g2l(&B[(size_t)(kn + br1) * N + n0 + bc1], &Bs[nxt][br1 * 128 + bc1]);
      if (step + 2 < nstep) {  // global_prefetch_b8 of the tile after next
        __builtin_prefetch(&A[(size_t)(m0 + ar0) * K + k0 + 64 + ac0], 0, 1);
        __builtin_prefetch(&B[(size_t)(k0 + 64 + br0) * N + n0 + bc0], 0, 1);
      }
    }

    // A fragment: lane holds row m, K cols {kb..kb+7, 16+kb..16+kb+7}
    Frag16 fa[2], fb[4];
#pragma unroll
    for (int rt = 0; rt < 2; ++rt) {
      int rr = wm * 32 + rt * 16 + mfr;
      fa[rt].u[0] = *reinterpret_cast<const uint4*>(&As[cur][rr * 32 + kb]);
      fa[rt].u[1] = *reinterpret_cast<const uint4*>(&As[cur][rr * 32 + kb + 16]);
    }
    // B fragment: lane holds column n = lane&15, K rows {kb..kb+7, 16+kb..16+kb+7}
#pragma unroll
    for (int ct = 0; ct < 4; ++ct) {
      const u16* p = &Bs[cur][wn * 64 + ct * 16 + mfr];
      u16* e = reinterpret_cast<u16*>(&fb[ct]);
#pragma unroll
      for (int s = 0; s < 8; ++s) {
        e[s]     = p[(kb + s) * 128];
        e[s + 8] = p[(kb + 16 + s) * 128];
      }
    }
#pragma unroll
    for (int rt = 0; rt < 2; ++rt)
#pragma unroll
      for (int ct = 0; ct < 4; ++ct)
        acc[rt][ct].v = __builtin_amdgcn_wmma_f32_16x16x32_bf16(
            false, fa[rt].v, false, fb[ct].v, (short)0, acc[rt][ct].v,
            false, false);

    wait_async_copies();   // our wave's async stores to LDS have landed
    __syncthreads();       // publish across waves before next step reads
  }

  // Epilogue: C tile layout row = r + 8*(lane>=16), col = lane&15
  const int rbase = m0 + wm * 32 + ((lane >= 16) ? 8 : 0);
  const int cbase = n0 + wn * 64 + (lane & 15);
#pragma unroll
  for (int rt = 0; rt < 2; ++rt)
#pragma unroll
    for (int ct = 0; ct < 4; ++ct) {
      int   cc = cbase + ct * 16;
      float bv = bias ? bias[cc] : 0.f;
#pragma unroll
      for (int r = 0; r < 8; ++r) {
        int   rr = rbase + rt * 16 + r;
        float v  = acc[rt][ct].f[r] + bv;
        if (relu) v = fmaxf(v, 0.f);
        if (C16) C16[(size_t)rr * N + cc] = f32_to_bf16(v);
        else     C32[(size_t)rr * N + cc] = v;
      }
    }
}

// ---------------------------------------------------------------------------
// Flash attention: one wave per (b,h, 16 query rows). Keys in blocks of 32.
// Q frags held in regs; K^T frags contiguous from KbT; V frags contiguous.
// Online softmax on WMMA C-tile layout (rows live per-VGPR, 16-lane halves).
// ---------------------------------------------------------------------------
__global__ __launch_bounds__(32) void attn_kernel(
    const u16* __restrict__ Qb, const u16* __restrict__ KbT,
    const u16* __restrict__ Vb, u16* __restrict__ Ob) {
  __shared__ u16 Ps[16 * 32];                 // P bounce buffer (A-frag relayout)
  const int   lane  = threadIdx.x;
  const int   q0    = blockIdx.x * 16;
  const int   bh    = blockIdx.y;             // b*8 + h
  const int   b     = bh >> 3, h = bh & 7;
  const int   m     = lane & 15;
  const int   kb    = (lane >= 16) ? 8 : 0;
  const int   hb    = (lane >= 16) ? 8 : 0;
  const float scale = 0.044194173824159216f;  // 1/sqrt(512) (reference uses D)

  Frag16 fq[2];
  {
    const u16* qrow = Qb + (size_t)(b * 1024 + q0 + m) * 512 + h * 64;
#pragma unroll
    for (int j = 0; j < 2; ++j) {
      fq[j].u[0] = *reinterpret_cast<const uint4*>(qrow + j * 32 + kb);
      fq[j].u[1] = *reinterpret_cast<const uint4*>(qrow + j * 32 + kb + 16);
    }
  }

  AccF  o[4];
  float mrow[8], lrow[8];
#pragma unroll
  for (int r = 0; r < 8; ++r) { mrow[r] = -1e30f; lrow[r] = 0.f; }
#pragma unroll
  for (int t = 0; t < 4; ++t)
#pragma unroll
    for (int r = 0; r < 8; ++r) o[t].f[r] = 0.f;

  for (int jb = 0; jb < 1024; jb += 32) {
    // S = scale * Q @ K^T  (two 16x16 tiles, K-dim 64 = 2 WMMA each)
    AccF s[2];
#pragma unroll
    for (int t = 0; t < 2; ++t) {
#pragma unroll
      for (int r = 0; r < 8; ++r) s[t].f[r] = 0.f;
#pragma unroll
      for (int j = 0; j < 2; ++j) {
        Frag16 fk;
        const u16* kp =
            KbT + (size_t)(bh * 64 + j * 32) * 1024 + jb + t * 16 + m;
        u16* ek = reinterpret_cast<u16*>(&fk);
#pragma unroll
        for (int s = 0; s < 8; ++s) {
          ek[s]     = kp[(size_t)(kb + s) * 1024];
          ek[s + 8] = kp[(size_t)(kb + 16 + s) * 1024];
        }
        s[t].v = __builtin_amdgcn_wmma_f32_16x16x32_bf16(
            false, fq[j].v, false, fk.v, (short)0, s[t].v, false, false);
      }
#pragma unroll
      for (int r = 0; r < 8; ++r) s[t].f[r] *= scale;
    }

    // Online softmax: row-wise max/sum via xor-shuffles within 16-lane halves
    float rm[8];
#pragma unroll
    for (int r = 0; r < 8; ++r) rm[r] = fmaxf(s[0].f[r], s[1].f[r]);
#pragma unroll
    for (int off = 8; off >= 1; off >>= 1)
#pragma unroll
      for (int r = 0; r < 8; ++r) rm[r] = fmaxf(rm[r], __shfl_xor(rm[r], off, 32));

    float mnew[8], al[8], rs[8];
#pragma unroll
    for (int r = 0; r < 8; ++r) {
      mnew[r] = fmaxf(mrow[r], rm[r]);
      al[r]   = __expf(mrow[r] - mnew[r]);
      rs[r]   = 0.f;
    }
#pragma unroll
    for (int t = 0; t < 2; ++t)
#pragma unroll
      for (int r = 0; r < 8; ++r) {
        float p = __expf(s[t].f[r] - mnew[r]);
        rs[r] += p;
        Ps[(r + hb) * 32 + t * 16 + (lane & 15)] = f32_to_bf16(p);
      }
#pragma unroll
    for (int off = 8; off >= 1; off >>= 1)
#pragma unroll
      for (int r = 0; r < 8; ++r) rs[r] += __shfl_xor(rs[r], off, 32);
#pragma unroll
    for (int r = 0; r < 8; ++r) {
      lrow[r] = lrow[r] * al[r] + rs[r];
      mrow[r] = mnew[r];
    }
#pragma unroll
    for (int t = 0; t < 4; ++t)
#pragma unroll
      for (int r = 0; r < 8; ++r) o[t].f[r] *= al[r];

    __syncthreads();
    Frag16 fp;   // P as A-fragment (16x32 over keys)
    fp.u[0] = *reinterpret_cast<const uint4*>(&Ps[m * 32 + kb]);
    fp.u[1] = *reinterpret_cast<const uint4*>(&Ps[m * 32 + 16 + kb]);

#pragma unroll
    for (int t2 = 0; t2 < 4; ++t2) {
      Frag16 fv;   // V column d = lane&15, key rows {kb..kb+7, 16+kb..16+kb+7}
      const u16* vp =
          Vb + (size_t)(b * 1024 + jb) * 512 + h * 64 + t2 * 16 + m;
      u16* ev = reinterpret_cast<u16*>(&fv);
#pragma unroll
      for (int s = 0; s < 8; ++s) {
        ev[s]     = vp[(size_t)(kb + s) * 512];
        ev[s + 8] = vp[(size_t)(kb + 16 + s) * 512];
      }
      o[t2].v = __builtin_amdgcn_wmma_f32_16x16x32_bf16(
          false, fp.v, false, fv.v, (short)0, o[t2].v, false, false);
    }
    __syncthreads();
  }

#pragma unroll
  for (int t2 = 0; t2 < 4; ++t2)
#pragma unroll
    for (int r = 0; r < 8; ++r) {
      float v = o[t2].f[r] / lrow[r];
      Ob[(size_t)(b * 1024 + q0 + r + hb) * 512 + h * 64 + t2 * 16 +
         (lane & 15)] = f32_to_bf16(v);
    }
}

// ---------------------------------------------------------------------------
// Fused residual-add + LayerNorm over D=512; emits fp32 and optional bf16.
// One 256-thread block per row.
// ---------------------------------------------------------------------------
__global__ __launch_bounds__(256) void add_ln_kernel(
    const float* __restrict__ resid, const float* __restrict__ yv,
    const float* __restrict__ g, const float* __restrict__ bt,
    float* __restrict__ out32, u16* __restrict__ out16) {
  __shared__ float red[16];
  const int    tid  = threadIdx.x;
  const size_t base = (size_t)blockIdx.x * 512;

  float v0 = resid[base + tid] + yv[base + tid];
  float v1 = resid[base + tid + 256] + yv[base + tid + 256];
  float s = v0 + v1, ss = v0 * v0 + v1 * v1;
#pragma unroll
  for (int off = 16; off >= 1; off >>= 1) {
    s  += __shfl_xor(s, off, 32);
    ss += __shfl_xor(ss, off, 32);
  }
  const int wave = tid >> 5, lane = tid & 31;
  if (lane == 0) { red[wave] = s; red[8 + wave] = ss; }
  __syncthreads();
  s = 0.f; ss = 0.f;
#pragma unroll
  for (int i = 0; i < 8; ++i) { s += red[i]; ss += red[8 + i]; }

  float mu   = s * (1.f / 512.f);
  float var  = ss * (1.f / 512.f) - mu * mu;
  float rstd = rsqrtf(var + 1e-5f);
  float o0   = (v0 - mu) * rstd * g[tid] + bt[tid];
  float o1   = (v1 - mu) * rstd * g[tid + 256] + bt[tid + 256];
  out32[base + tid]       = o0;
  out32[base + tid + 256] = o1;
  if (out16) {
    out16[base + tid]       = f32_to_bf16(o0);
    out16[base + tid + 256] = f32_to_bf16(o1);
  }
}

// ---------------------------------------------------------------------------
// Host orchestration
// ---------------------------------------------------------------------------
extern "C" void kernel_launch(void* const* d_in, const int* in_sizes, int n_in,
                              void* d_out, int out_size, void* d_ws,
                              size_t ws_size, hipStream_t stream) {
  (void)in_sizes; (void)n_in; (void)out_size; (void)ws_size;
  const float* x    = (const float*)d_in[0];
  const float* Wq   = (const float*)d_in[1];
  const float* bq   = (const float*)d_in[2];
  const float* Wk   = (const float*)d_in[3];
  const float* bk   = (const float*)d_in[4];
  const float* Wv   = (const float*)d_in[5];
  const float* bv   = (const float*)d_in[6];
  const float* Wo   = (const float*)d_in[7];
  const float* bo   = (const float*)d_in[8];
  const float* c1w  = (const float*)d_in[9];
  const float* c1b  = (const float*)d_in[10];
  const float* ln1g = (const float*)d_in[11];
  const float* ln1b = (const float*)d_in[12];
  const float* c2w  = (const float*)d_in[13];
  const float* c2b  = (const float*)d_in[14];
  const float* ln2g = (const float*)d_in[15];
  const float* ln2b = (const float*)d_in[16];
  float* out = (float*)d_out;

  constexpr size_t SZ_XB  = (size_t)8192 * 512 * 2;    // bf16 activations
  constexpr size_t SZ_WT  = (size_t)512 * 512 * 2;     // 512x512 bf16 weight
  constexpr size_t SZ_C1T = (size_t)512 * 2048 * 2;    // FFN weight bf16
  constexpr size_t SZ_F32 = (size_t)8192 * 512 * 4;    // fp32 activations

  char* ws = (char*)d_ws;
  size_t off = 0;
  u16*   xb   = (u16*)(ws + off); off += SZ_XB;
  u16*   WqT  = (u16*)(ws + off); off += SZ_WT;
  u16*   WkT  = (u16*)(ws + off); off += SZ_WT;
  u16*   WvT  = (u16*)(ws + off); off += SZ_WT;
  u16*   WoT  = (u16*)(ws + off); off += SZ_WT;
  u16*   c1T  = (u16*)(ws + off); off += SZ_C1T;
  u16*   c2T  = (u16*)(ws + off); off += SZ_C1T;
  u16*   Qb   = (u16*)(ws + off); off += SZ_XB;
  u16*   Kb   = (u16*)(ws + off); off += SZ_XB;
  u16*   KbT  = (u16*)(ws + off); off += SZ_XB;
  u16*   Vb   = (u16*)(ws + off); off += SZ_XB;
  u16*   attnb= (u16*)(ws + off); off += SZ_XB;
  float* oproj= (float*)(ws + off); off += SZ_F32;
  float* x1   = (float*)(ws + off); off += SZ_F32;
  u16*   x1b  = (u16*)(ws + off); off += SZ_XB;
  u16*   yb   = Qb;      // reuse dead Q/K/KbT/V region (exactly 32 MiB)
  float* ffn2 = oproj;   // reuse dead O-projection buffer

  const int NX = 8192 * 512;
  // 1) convert x, weights (weights -> [K,N] bf16)
  cvt_bf16_kernel<<<(NX + 255) / 256, 256, 0, stream>>>(x, xb, NX);
  tpose_cvt_kernel<<<(512 * 512 + 255) / 256, 256, 0, stream>>>(Wq, WqT, 512, 512);
  tpose_cvt_kernel<<<(512 * 512 + 255) / 256, 256, 0, stream>>>(Wk, WkT, 512, 512);
  tpose_cvt_kernel<<<(512 * 512 + 255) / 256, 256, 0, stream>>>(Wv, WvT, 512, 512);
  tpose_cvt_kernel<<<(512 * 512 + 255) / 256, 256, 0, stream>>>(Wo, WoT, 512, 512);
  tpose_cvt_kernel<<<(2048 * 512 + 255) / 256, 256, 0, stream>>>(c1w, c1T, 2048, 512);
  tpose_cvt_kernel<<<(512 * 2048 + 255) / 256, 256, 0, stream>>>(c2w, c2T, 512, 2048);

  // 2) Q,K,V projections (bf16 out)
  dim3 g512(4, 64), blk(256);
  gemm_bf16_kernel<<<g512, blk, 0, stream>>>(xb, WqT, bq, nullptr, Qb, 8192, 512, 512, 0);
  gemm_bf16_kernel<<<g512, blk, 0, stream>>>(xb, WkT, bk, nullptr, Kb, 8192, 512, 512, 0);
  gemm_bf16_kernel<<<g512, blk, 0, stream>>>(xb, WvT, bv, nullptr, Vb, 8192, 512, 512, 0);
  kt_tpose_kernel<<<(NX + 255) / 256, 256, 0, stream>>>(Kb, KbT);

  // 3) attention -> attnb (bf16)
  attn_kernel<<<dim3(64, 64), 32, 0, stream>>>(Qb, KbT, Vb, attnb);

  // 4) output projection (fp32) + residual LN
  gemm_bf16_kernel<<<g512, blk, 0, stream>>>(attnb, WoT, bo, oproj, nullptr, 8192, 512, 512, 0);
  add_ln_kernel<<<8192, 256, 0, stream>>>(x, oproj, ln1g, ln1b, x1, x1b);

  // 5) FFN: relu(x1 @ c1^T) -> yb (bf16), then @ c2^T -> fp32, LN -> out
  gemm_bf16_kernel<<<dim3(16, 64), blk, 0, stream>>>(x1b, c1T, c1b, nullptr, yb, 8192, 2048, 512, 1);
  gemm_bf16_kernel<<<g512, blk, 0, stream>>>(yb, c2T, c2b, ffn2, nullptr, 8192, 512, 2048, 0);
  add_ln_kernel<<<8192, 256, 0, stream>>>(x1, ffn2, ln2g, ln2b, out, nullptr);
}
